// SimilarityAdj_61649960567143
// MI455X (gfx1250) — compile-verified, matrix operand order checked by reading, not run on hardware
//
#include <hip/hip_runtime.h>
#include <stdint.h>

// ---------------- CDNA5 types ----------------
typedef __attribute__((ext_vector_type(16))) __bf16        v16bf;
typedef __attribute__((ext_vector_type(8)))  float         v8f;
typedef __attribute__((ext_vector_type(4)))  unsigned int  v4u;
typedef __attribute__((ext_vector_type(4)))  float         v4f;

static constexpr int Bn = 8;
static constexpr int Nn = 2048;
static constexpr int Dn = 512;
static constexpr int KTOP = 10;

// Assemble one 16-bit WMMA operand (A: 16x32 rows-in-lanes, or B: 32x16
// cols-in-lanes) from a row-major bf16 array. Per ISA 7.12.2, for lane group
// h = lane>>4: VGPR j<4 holds K = h*8 + 2j (+1), VGPR j>=4 holds K = 16 + h*8
// + 2(j-4) (+1)  => two contiguous 16-byte quads per operand.
static __device__ __forceinline__ v16bf load_op(const __bf16* rowbase, int kbase, int h) {
  union { v16bf v; v4u q[2]; } u;
  u.q[0] = *(const v4u*)(rowbase + kbase + h * 8);
  u.q[1] = *(const v4u*)(rowbase + kbase + 16 + h * 8);
  return u.v;
}

static __device__ __forceinline__ v8f wmma_bf16(v16bf a, v16bf b, v8f c) {
  return __builtin_amdgcn_wmma_f32_16x16x32_bf16(false, a, false, b, (short)0, c, false, false);
}

// ---------------- 1) pack weights: W (DxD f32) -> W^T hi/lo bf16 ----------------
__global__ void pack_w_kernel(const float* __restrict__ W0, const float* __restrict__ W1,
                              unsigned short* W0h_, unsigned short* W0l_,
                              unsigned short* W1h_, unsigned short* W1l_) {
  int tid = blockIdx.x * blockDim.x + threadIdx.x;       // 2*512*512 threads
  int which = tid >> 18;
  int e = tid & ((1 << 18) - 1);
  int k = e >> 9, n = e & 511;
  const float* W = which ? W1 : W0;
  float x = W[k * Dn + n];
  __bf16 hi = (__bf16)x;
  __bf16 lo = (__bf16)(x - (float)hi);
  __bf16* Wh = (__bf16*)(which ? W1h_ : W0h_);
  __bf16* Wl = (__bf16*)(which ? W1l_ : W0l_);
  Wh[n * Dn + k] = hi;
  Wl[n * Dn + k] = lo;
}

// ---------------- 2) pack X -> hi/lo bf16, and aa = sum(x^2) ----------------
__global__ void pack_x_kernel(const float* __restrict__ X,
                              unsigned short* Xh_, unsigned short* Xl_,
                              float* __restrict__ aa) {
  int lane = threadIdx.x & 31, w = threadIdx.x >> 5;
  int rid = blockIdx.x * 8 + w;                          // row in [0, B*N)
  const float* xr = X + (size_t)rid * Dn;
  __bf16* Xh = (__bf16*)Xh_;
  __bf16* Xl = (__bf16*)Xl_;
  float ss = 0.0f;
#pragma unroll
  for (int j = 0; j < Dn / 32; ++j) {
    int c = lane + j * 32;
    float x = xr[c];
    __bf16 hi = (__bf16)x;
    Xh[(size_t)rid * Dn + c] = hi;
    Xl[(size_t)rid * Dn + c] = (__bf16)(x - (float)hi);
    ss += x * x;
  }
#pragma unroll
  for (int off = 16; off >= 1; off >>= 1) ss += __shfl_xor(ss, off, 32);
  if (lane == 0) aa[rid] = ss;
}

// ---------------- 3) projection: theta = X W0, phi = X W1 (split-bf16 WMMA) ----------------
// Each wave computes a 16x32 strip of BOTH theta and phi: A loads amortized
// over 2 column subtiles and 2 weight matrices -> 20 b128 loads : 12 WMMAs.
__global__ __launch_bounds__(128) void proj_kernel(
    const unsigned short* Xh_, const unsigned short* Xl_,
    const unsigned short* W0h_, const unsigned short* W0l_,
    const unsigned short* W1h_, const unsigned short* W1l_,
    unsigned short* Th_, unsigned short* Tl_, unsigned short* Ph_, unsigned short* Pl_) {
  const __bf16* Xh = (const __bf16*)Xh_;  const __bf16* Xl = (const __bf16*)Xl_;
  const __bf16* W0h = (const __bf16*)W0h_; const __bf16* W0l = (const __bf16*)W0l_;
  const __bf16* W1h = (const __bf16*)W1h_; const __bf16* W1l = (const __bf16*)W1l_;
  __bf16* Th = (__bf16*)Th_; __bf16* Tl = (__bf16*)Tl_;
  __bf16* Ph = (__bf16*)Ph_; __bf16* Pl = (__bf16*)Pl_;

  int lane = threadIdx.x & 31, w = threadIdx.x >> 5;
  int rl = lane & 15, h = lane >> 4;
  int b = blockIdx.y;
  int rowTile = blockIdx.x >> 2;                 // N/16 = 128 row tiles
  int colGrp = blockIdx.x & 3;                   // D/128 = 4 col groups
  int r0 = rowTile * 16;
  int c0 = colGrp * 128 + w * 32;                // 32 columns per wave

  const __bf16* Ah = Xh + (size_t)(b * Nn + r0 + rl) * Dn;
  const __bf16* Al = Xl + (size_t)(b * Nn + r0 + rl) * Dn;
  const __bf16* B0h[2], * B0l[2], * B1h[2], * B1l[2];
#pragma unroll
  for (int s = 0; s < 2; ++s) {
    B0h[s] = W0h + (size_t)(c0 + s * 16 + rl) * Dn;
    B0l[s] = W0l + (size_t)(c0 + s * 16 + rl) * Dn;
    B1h[s] = W1h + (size_t)(c0 + s * 16 + rl) * Dn;
    B1l[s] = W1l + (size_t)(c0 + s * 16 + rl) * Dn;
  }

  v8f accT[2] = {{}, {}};
  v8f accP[2] = {{}, {}};
  for (int kb = 0; kb < Dn; kb += 32) {
    v16bf ah = load_op(Ah, kb, h), al = load_op(Al, kb, h);
#pragma unroll
    for (int s = 0; s < 2; ++s) {
      v16bf bh0 = load_op(B0h[s], kb, h), bl0 = load_op(B0l[s], kb, h);
      v16bf bh1 = load_op(B1h[s], kb, h), bl1 = load_op(B1l[s], kb, h);
      accT[s] = wmma_bf16(ah, bh0, accT[s]);
      accT[s] = wmma_bf16(ah, bl0, accT[s]);
      accT[s] = wmma_bf16(al, bh0, accT[s]);
      accP[s] = wmma_bf16(ah, bh1, accP[s]);
      accP[s] = wmma_bf16(ah, bl1, accP[s]);
      accP[s] = wmma_bf16(al, bh1, accP[s]);
    }
  }
#pragma unroll
  for (int s = 0; s < 2; ++s) {
#pragma unroll
    for (int i = 0; i < 8; ++i) {
      int row = r0 + i + h * 8;
      int col = c0 + s * 16 + rl;
      size_t o = (size_t)(b * Nn + row) * Dn + col;
      float t = accT[s][i]; __bf16 thv = (__bf16)t; Th[o] = thv; Tl[o] = (__bf16)(t - (float)thv);
      float p = accP[s][i]; __bf16 phv = (__bf16)p; Ph[o] = phv; Pl[o] = (__bf16)(p - (float)phv);
    }
  }
}

// ---------------- 4) deterministic row norms of theta/phi ----------------
__global__ void rownorm_kernel(const unsigned short* Th_, const unsigned short* Tl_,
                               const unsigned short* Ph_, const unsigned short* Pl_,
                               float* __restrict__ tn2, float* __restrict__ pn2) {
  const __bf16* Th = (const __bf16*)Th_; const __bf16* Tl = (const __bf16*)Tl_;
  const __bf16* Ph = (const __bf16*)Ph_; const __bf16* Pl = (const __bf16*)Pl_;
  int lane = threadIdx.x & 31, w = threadIdx.x >> 5;
  int rid = blockIdx.x * 8 + w;
  float st = 0.0f, sp = 0.0f;
#pragma unroll
  for (int j = 0; j < Dn / 32; ++j) {
    size_t o = (size_t)rid * Dn + lane + j * 32;
    float t = (float)Th[o] + (float)Tl[o]; st += t * t;
    float p = (float)Ph[o] + (float)Pl[o]; sp += p * p;
  }
#pragma unroll
  for (int off = 16; off >= 1; off >>= 1) { st += __shfl_xor(st, off, 32); sp += __shfl_xor(sp, off, 32); }
  if (lane == 0) { tn2[rid] = st; pn2[rid] = sp; }
}

// ---------------- 5) fused sim + dist + double softmax + top-K -> selected ----------------
// 256 threads (8 waves) per block; each wave computes 16 rows x 256 cols in
// 4 passes of a 16x64 register tile (A loads amortized over 4 subtiles ->
// 40 b128 loads : 24 WMMAs). 256 KB LDS holds the sim & dist row-block.
__global__ __launch_bounds__(256) void simtopk_kernel(
    const unsigned short* Th_, const unsigned short* Tl_,
    const unsigned short* Ph_, const unsigned short* Pl_,
    const unsigned short* Xh_, const unsigned short* Xl_,
    const float* __restrict__ tn2, const float* __restrict__ pn2,
    const float* __restrict__ aa, const int* __restrict__ seq_len,
    int* __restrict__ selected) {
  extern __shared__ float lds[];                  // 16*2048 sim | 16*2048 dist = 256 KB
  float* Ssim = lds;
  float* Sdis = lds + 16 * Nn;

  const __bf16* Th = (const __bf16*)Th_; const __bf16* Tl = (const __bf16*)Tl_;
  const __bf16* Ph = (const __bf16*)Ph_; const __bf16* Pl = (const __bf16*)Pl_;
  const __bf16* Xh = (const __bf16*)Xh_; const __bf16* Xl = (const __bf16*)Xl_;

  int lane = threadIdx.x & 31, w = threadIdx.x >> 5;   // 8 waves
  int rl = lane & 15, h = lane >> 4;
  int b = blockIdx.y;
  int r0 = blockIdx.x * 16;
  const size_t rb = (size_t)b * Nn;

  const __bf16* ATh = Th + (rb + r0 + rl) * Dn;
  const __bf16* ATl = Tl + (rb + r0 + rl) * Dn;
  const __bf16* AXh = Xh + (rb + r0 + rl) * Dn;
  const __bf16* AXl = Xl + (rb + r0 + rl) * Dn;

  // ---- phase 1: this wave computes 16 rows x 256 cols of sim & dist ----
  for (int p = 0; p < 4; ++p) {
    int c0p = w * 256 + p * 64;                  // 64-column register tile
    const __bf16* BPh[4], * BPl[4], * BXh[4], * BXl[4];
#pragma unroll
    for (int s = 0; s < 4; ++s) {
      BPh[s] = Ph + (rb + c0p + s * 16 + rl) * Dn;
      BPl[s] = Pl + (rb + c0p + s * 16 + rl) * Dn;
      BXh[s] = Xh + (rb + c0p + s * 16 + rl) * Dn;
      BXl[s] = Xl + (rb + c0p + s * 16 + rl) * Dn;
    }
    v8f asim[4] = {{}, {}, {}, {}};
    v8f ag[4] = {{}, {}, {}, {}};
    for (int kb = 0; kb < Dn; kb += 32) {
      v16bf ath = load_op(ATh, kb, h), atl = load_op(ATl, kb, h);
      v16bf axh = load_op(AXh, kb, h), axl = load_op(AXl, kb, h);
#pragma unroll
      for (int s = 0; s < 4; ++s) {
        v16bf bph = load_op(BPh[s], kb, h), bpl = load_op(BPl[s], kb, h);
        v16bf bxh = load_op(BXh[s], kb, h), bxl = load_op(BXl[s], kb, h);
        asim[s] = wmma_bf16(ath, bph, asim[s]);
        asim[s] = wmma_bf16(ath, bpl, asim[s]);
        asim[s] = wmma_bf16(atl, bph, asim[s]);
        ag[s] = wmma_bf16(axh, bxh, ag[s]);
        ag[s] = wmma_bf16(axh, bxl, ag[s]);
        ag[s] = wmma_bf16(axl, bxh, ag[s]);
      }
    }
#pragma unroll
    for (int s = 0; s < 4; ++s) {
      int col = c0p + s * 16 + rl;
      float pn = sqrtf(pn2[rb + col]);
      float aac = aa[rb + col];
#pragma unroll
      for (int i = 0; i < 8; ++i) {
        int rloc = i + h * 8;
        int rowg = r0 + rloc;
        float tn = sqrtf(tn2[rb + rowg]);
        Ssim[rloc * Nn + col] = asim[s][i] / (tn * pn + 1e-20f);
        float d2 = aa[rb + rowg] + aac - 2.0f * ag[s][i];
        Sdis[rloc * Nn + col] = sqrtf(fmaxf(d2, 0.0f));
      }
    }
  }
  __syncthreads();

  // ---- phase 2: per-row softmaxes, combine, top-K (value desc, index desc) ----
  int L = seq_len[b];
  for (int rr = 0; rr < 2; ++rr) {
    int rloc = w * 2 + rr;
    int rowg = r0 + rloc;
    float* srow = Ssim + rloc * Nn;
    float* drow = Sdis + rloc * Nn;

    float m1 = -__builtin_inff(), m2 = -__builtin_inff();
    for (int c = lane; c < Nn; c += 32)
      if (c < L) { m1 = fmaxf(m1, srow[c]); m2 = fmaxf(m2, drow[c]); }
#pragma unroll
    for (int off = 16; off >= 1; off >>= 1) {
      m1 = fmaxf(m1, __shfl_xor(m1, off, 32));
      m2 = fmaxf(m2, __shfl_xor(m2, off, 32));
    }
    float s1 = 0.0f, s2 = 0.0f;
    for (int c = lane; c < Nn; c += 32)
      if (c < L) { s1 += expf(srow[c] - m1); s2 += expf(drow[c] - m2); }
#pragma unroll
    for (int off = 16; off >= 1; off >>= 1) {
      s1 += __shfl_xor(s1, off, 32);
      s2 += __shfl_xor(s2, off, 32);
    }
    float i1 = 1.0f / s1, i2 = 0.5f / s2;
    bool rowok = rowg < L;
    for (int c = lane; c < Nn; c += 32) {
      float Sv = 0.0f;
      if (rowok && c < L && c != rowg)
        Sv = expf(srow[c] - m1) * i1 + expf(drow[c] - m2) * i2;
      srow[c] = Sv;                               // reuse sim row as S row
    }
    int top[KTOP];
#pragma unroll
    for (int tsel = 0; tsel < KTOP; ++tsel) {
      float bv = -__builtin_inff(); int bi = -1;
      for (int c = lane; c < Nn; c += 32) {
        float v = srow[c];
        if (v > bv || (v == bv && c > bi)) { bv = v; bi = c; }
      }
#pragma unroll
      for (int off = 16; off >= 1; off >>= 1) {
        float ov = __shfl_xor(bv, off, 32);
        int oi = __shfl_xor(bi, off, 32);
        if (ov > bv || (ov == bv && oi > bi)) { bv = ov; bi = oi; }
      }
      top[tsel] = bi;
      if (lane == (bi & 31)) srow[bi] = -__builtin_inff();
    }
    bool cont = false;
#pragma unroll
    for (int tsel = 0; tsel < KTOP; ++tsel) cont = cont || (top[tsel] == rowg);
    if (lane == 0) selected[rb + rowg] = cont ? top[KTOP - 1] : rowg;
  }
}

// ---------------- 6) histogram DV ----------------
__global__ void hist_kernel(const int* __restrict__ selected, int* __restrict__ DV) {
  int i = blockIdx.x * blockDim.x + threadIdx.x;
  if (i < Bn * Nn) {
    int b = i >> 11;
    atomicAdd(&DV[(b << 11) + selected[i]], 1);
  }
}

// ---------------- 7) G = dv2 * M * dv2 (diagonal counts, IEEE inf*0 = NaN) ----------------
__global__ void writeG_kernel(const int* __restrict__ DV, float* __restrict__ G) {
  size_t t = (size_t)blockIdx.x * blockDim.x + threadIdx.x;   // B*N*N/4 threads
  int w4 = (int)(t & (Nn / 4 - 1));
  size_t r = t >> 9;
  int v = (int)(r & (Nn - 1));
  int b = (int)(r >> 11);
  int dvv = DV[(b << 11) + v];
  float rv = (dvv > 0) ? 1.0f / sqrtf((float)dvv) : __builtin_inff();
  v4f o;
  int wbase = w4 * 4;
#pragma unroll
  for (int j = 0; j < 4; ++j) {
    int wc = wbase + j;
    int dw = DV[(b << 11) + wc];
    float rw = (dw > 0) ? 1.0f / sqrtf((float)dw) : __builtin_inff();
    float m = (v == wc) ? (float)dvv : 0.0f;
    o[j] = (rv * m) * rw;
  }
  *(v4f*)(G + ((size_t)b * Nn * Nn + (size_t)v * Nn + wbase)) = o;
}

// ---------------- launcher ----------------
extern "C" void kernel_launch(void* const* d_in, const int* in_sizes, int n_in,
                              void* d_out, int out_size, void* d_ws, size_t ws_size,
                              hipStream_t stream) {
  const float* X = (const float*)d_in[0];
  const float* W0 = (const float*)d_in[1];
  const float* W1 = (const float*)d_in[2];
  const int* seq = (const int*)d_in[3];
  float* out = (float*)d_out;

  char* p = (char*)d_ws;
  auto alloc = [&](size_t bytes) -> char* {
    char* q = p;
    p += (bytes + 255) & ~(size_t)255;
    return q;
  };
  const size_t szW = (size_t)Dn * Dn * 2;               // 512 KB
  const size_t szX = (size_t)Bn * Nn * Dn * 2;          // 16 MB
  const size_t szRow = (size_t)Bn * Nn * 4;             // 64 KB

  unsigned short* W0h = (unsigned short*)alloc(szW);
  unsigned short* W0l = (unsigned short*)alloc(szW);
  unsigned short* W1h = (unsigned short*)alloc(szW);
  unsigned short* W1l = (unsigned short*)alloc(szW);
  unsigned short* Xh = (unsigned short*)alloc(szX);
  unsigned short* Xl = (unsigned short*)alloc(szX);
  float* aa = (float*)alloc(szRow);
  unsigned short* Th = (unsigned short*)alloc(szX);
  unsigned short* Tl = (unsigned short*)alloc(szX);
  unsigned short* Ph = (unsigned short*)alloc(szX);
  unsigned short* Pl = (unsigned short*)alloc(szX);
  float* tn2 = (float*)alloc(szRow);
  float* pn2 = (float*)alloc(szRow);
  int* sel = (int*)alloc(szRow);
  int* DV = (int*)alloc(szRow);

  pack_w_kernel<<<2 * Dn * Dn / 256, 256, 0, stream>>>(W0, W1, W0h, W0l, W1h, W1l);
  pack_x_kernel<<<Bn * Nn / 8, 256, 0, stream>>>(X, Xh, Xl, aa);
  proj_kernel<<<dim3((Nn / 16) * (Dn / 128), Bn), 128, 0, stream>>>(
      Xh, Xl, W0h, W0l, W1h, W1l, Th, Tl, Ph, Pl);
  rownorm_kernel<<<Bn * Nn / 8, 256, 0, stream>>>(Th, Tl, Ph, Pl, tn2, pn2);
  simtopk_kernel<<<dim3(Nn / 16, Bn), 256, 2u * 16u * Nn * sizeof(float), stream>>>(
      Th, Tl, Ph, Pl, Xh, Xl, tn2, pn2, aa, seq, sel);
  hipMemsetAsync(DV, 0, szRow, stream);
  hist_kernel<<<(Bn * Nn + 255) / 256, 256, 0, stream>>>(sel, DV);
  writeG_kernel<<<(int)(((size_t)Bn * Nn * Nn / 4) / 256), 256, 0, stream>>>(DV, out);
}
